// model_5566277616413
// MI455X (gfx1250) — matrix operand order repaired and checked by reading, not compiled, not run on hardware
//
#include <hip/hip_runtime.h>
#include <hip/hip_bf16.h>
#include <math.h>

#define HSZ 512
#define H2  256
#define BB  8
#define MAXLEN 48
#define VSZ 50000
#define NVSZ 20000
#define VVSZ 10000
#define SLEN 512

typedef __bf16 bf16_t;
typedef __attribute__((ext_vector_type(16))) __bf16 v16bf;
typedef __attribute__((ext_vector_type(8)))  __bf16 v8bf;
typedef __attribute__((ext_vector_type(8)))  float  v8f;

__device__ __forceinline__ bf16_t f2bf(float f) {
  union { float f; unsigned u; } v; v.f = f;
  unsigned r = (v.u + 0x7FFFu + ((v.u >> 16) & 1u)) >> 16;
  unsigned short s = (unsigned short)r;
  bf16_t o; __builtin_memcpy(&o, &s, sizeof(o));
  return o;
}
__device__ __forceinline__ float sigmoidf_(float x){ return 1.0f/(1.0f+__expf(-x)); }

// ---------------- weight/activation conversion ----------------
__global__ void k_f2bf(const float* __restrict__ src, bf16_t* __restrict__ dst, int n) {
  int i = blockIdx.x*256 + threadIdx.x;
  if (i < n) dst[i] = f2bf(src[i]);
}

__global__ void k_embed_bf(const float* __restrict__ table, const int* __restrict__ ids,
                           bf16_t* __restrict__ dst, int rows, int width) {
  int i = blockIdx.x*256 + threadIdx.x;
  if (i < rows*width) {
    int r = i / width, c = i % width;
    dst[i] = f2bf(table[(size_t)ids[r]*width + c]);
  }
}

// pack [rows<=8, L] f32 -> [16, L] bf16 zero-padded
__global__ void k_pack1(const float* __restrict__ s, bf16_t* __restrict__ A, int L) {
  int i = blockIdx.x*256 + threadIdx.x;
  if (i < 16*L) { int m = i/L, k = i%L; A[i] = (m < BB) ? f2bf(s[m*L+k]) : f2bf(0.0f); }
}
__global__ void k_pack2(const float* __restrict__ s0, int L0, const float* __restrict__ s1, int L1,
                        bf16_t* __restrict__ A) {
  int L = L0 + L1;
  int i = blockIdx.x*256 + threadIdx.x;
  if (i < 16*L) {
    int m = i/L, k = i%L; float v = 0.0f;
    if (m < BB) v = (k < L0) ? s0[m*L0+k] : s1[m*L1 + (k-L0)];
    A[i] = f2bf(v);
  }
}
// decin = [decemb[prev] | ops | interstack]  -> [16, 2048] bf16
__global__ void k_pack_dec(const float* __restrict__ demb, const int* __restrict__ prev,
                           const float* __restrict__ ops, const float* __restrict__ istk,
                           bf16_t* __restrict__ A) {
  const int L = 4*HSZ;
  int i = blockIdx.x*256 + threadIdx.x;
  if (i < 16*L) {
    int m = i/L, k = i%L; float v = 0.0f;
    if (m < BB) {
      if (k < HSZ)        v = demb[(size_t)prev[m]*HSZ + k];
      else if (k < 2*HSZ) v = ops[m*HSZ + (k-HSZ)];
      else                v = istk[m*2*HSZ + (k-2*HSZ)];
    }
    A[i] = f2bf(v);
  }
}

// ---------------- WMMA bf16 GEMM:  D[M,N] = act(A[16*gy, K] @ W[N,K]^T + bias (+Cin)) ----------------
// block = 256 threads = 8 waves, wave w -> N-tile (blockIdx.x*8 + w), blockIdx.y -> M-tile.
// A tile staged into LDS with CDNA5 async global->LDS DMA (ASYNCcnt path).
__global__ void k_wmma_gemm(const bf16_t* __restrict__ A, const bf16_t* __restrict__ W,
                            const float* __restrict__ bias, const float* __restrict__ Cin,
                            float* __restrict__ D, int M, int N, int K, int act) {
  extern __shared__ bf16_t As[];                    // 16 x K tile of A (LDS offset 0)
  const int tid = threadIdx.x;
  const int mtile = blockIdx.y;
  const bf16_t* Ab = A + (size_t)mtile * 16 * K;
  const int nvec = (16*K) >> 3;                     // # of 16-byte chunks
  for (int i = tid; i < nvec; i += blockDim.x) {
    unsigned lofs = (unsigned)(i * 16);             // LDS byte offset (As is at offset 0)
    unsigned long long ga = (unsigned long long)(const void*)(((const v8bf*)Ab) + i);
    // async DMA: global memory -> LDS, no VGPR round-trip, tracked by ASYNCcnt
    asm volatile("global_load_async_to_lds_b128 %0, %1, off"
                 :: "v"(lofs), "v"(ga) : "memory");
  }
  asm volatile("s_wait_asynccnt 0x0" ::: "memory");
  __syncthreads();

  const int wave = tid >> 5;
  const int lane = tid & 31;
  const int ntile = blockIdx.x * 8 + wave;
  if (ntile * 16 >= N) return;
  const int nbase = ntile * 16;
  const int nl = lane & 15;
  const int n  = nbase + nl;
  const int khalf = (lane >> 4) * 8;                // A-frag K sub-pattern
  const int kboff = (lane >> 4) * 16;               // unused; B-frag uses khalf, same as A
  const bf16_t* arow = As + (size_t)nl * K;
  const bf16_t* wrow = W  + (size_t)n  * K;

  v8f acc;
  const int mb = (lane < 16) ? 0 : 8;
  if (Cin) {
#pragma unroll
    for (int v = 0; v < 8; v++) {
      int m = mtile*16 + mb + v;
      acc[v] = (m < M) ? Cin[(size_t)m*N + n] : 0.0f;
    }
  } else {
#pragma unroll
    for (int v = 0; v < 8; v++) acc[v] = 0.0f;
  }

  for (int k0 = 0; k0 < K; k0 += 32) {
    v8bf alo = *(const v8bf*)(arow + k0 + khalf);
    v8bf ahi = *(const v8bf*)(arow + k0 + khalf + 16);
    v8bf blo = *(const v8bf*)(wrow + k0 + khalf);
    v8bf bhi = *(const v8bf*)(wrow + k0 + khalf + 16);
    __builtin_prefetch(wrow + k0 + 64, 0, 3);       // WGP-scope prefetch of next weight tile
    v16bf af, bfv;
#pragma unroll
    for (int i = 0; i < 8; i++) { af[i]=alo[i]; af[8+i]=ahi[i]; bfv[i]=blo[i]; bfv[8+i]=bhi[i]; }
    acc = __builtin_amdgcn_wmma_f32_16x16x32_bf16(false, af, false, bfv, (short)0, acc, false, false);
  }

  float bv = bias ? bias[n] : 0.0f;
#pragma unroll
  for (int v = 0; v < 8; v++) {
    int m = mtile*16 + mb + v;
    if (m < M) {
      float x = acc[v] + bv;
      if (act == 1) x = tanhf(x);
      D[(size_t)m*N + n] = x;
    }
  }
}

// ---------------- encoder recurrent scan (blockIdx 0 = fwd, 1 = bwd) ----------------
__global__ void k_enc_scan(const float* __restrict__ Xf, const float* __restrict__ Xb,
                           const float* __restrict__ WhhF, const float* __restrict__ WhhB,
                           float* __restrict__ outF, float* __restrict__ outB,
                           float* __restrict__ hF, float* __restrict__ cF,
                           float* __restrict__ hB, float* __restrict__ cB) {
  __shared__ float hs[H2];
  const int t = threadIdx.x;                        // 256 threads
  const bool fwd = (blockIdx.x == 0);
  const float* X   = fwd ? Xf : Xb;
  const float* Whh = fwd ? WhhF : WhhB;
  float* out = fwd ? outF : outB;
  float c = 0.0f;
  hs[t] = 0.0f;
  __syncthreads();
  for (int s = 0; s < SLEN; s++) {
    int step = fwd ? s : (SLEN - 1 - s);
    const float* xg = X + (size_t)step * (4*H2);
    float gi = xg[t], gf = xg[H2+t], gg = xg[2*H2+t], go = xg[3*H2+t];
    const float* wi = Whh + (size_t)t        * H2;
    const float* wf = Whh + (size_t)(H2+t)   * H2;
    const float* wg = Whh + (size_t)(2*H2+t) * H2;
    const float* wo = Whh + (size_t)(3*H2+t) * H2;
    for (int k = 0; k < H2; k++) {
      float hk = hs[k];
      gi += hk * wi[k]; gf += hk * wf[k]; gg += hk * wg[k]; go += hk * wo[k];
    }
    float cn = sigmoidf_(gf) * c + sigmoidf_(gi) * tanhf(gg);
    float hn = sigmoidf_(go) * tanhf(cn);
    c = cn;
    __syncthreads();
    hs[t] = hn;
    out[(size_t)step * H2 + t] = hn;
    __syncthreads();
  }
  if (fwd) { hF[t] = hs[t]; cF[t] = c; } else { hB[t] = hs[t]; cB[t] = c; }
}

__global__ void k_build_enc(const float* __restrict__ outF, const float* __restrict__ outB,
                            float* __restrict__ enc, bf16_t* __restrict__ encb) {
  int i = blockIdx.x*256 + threadIdx.x;
  if (i < SLEN*HSZ) {
    int s = i / HSZ, j = i % HSZ;
    float v = (j < H2) ? outF[s*H2 + j] : outB[s*H2 + (j-H2)];
    enc[i] = v; encb[i] = f2bf(v);
  }
}

__global__ void k_init_states(const float* __restrict__ hF, const float* __restrict__ cF,
                              const float* __restrict__ hB, const float* __restrict__ cB,
                              float* hdec, float* cdec, float* hint, float* cint,
                              float* ops, int* prev, float* scores) {
  int i = blockIdx.x*256 + threadIdx.x;
  if (i < BB*HSZ) {
    int j = i % HSZ;
    float hv = (j < H2) ? hF[j] : hB[j-H2];
    float cv = (j < H2) ? cF[j] : cB[j-H2];
    hdec[i]=hv; cdec[i]=cv; hint[i]=hv; cint[i]=cv; ops[i]=0.0f;
  }
  if (i < BB) { prev[i] = 3; scores[i] = (i==0) ? 0.0f : -1e9f; }
}

// ---------------- elementwise / small kernels ----------------
__global__ void k_lstm_cell(const float* __restrict__ g, const float* __restrict__ cin,
                            float* __restrict__ hout, float* __restrict__ cout, int H) {
  int i = blockIdx.x*256 + threadIdx.x;
  if (i < BB*H) {
    int m = i / H, j = i % H;
    const float* gr = g + (size_t)m * 4 * H;
    float gi = gr[j], gf = gr[H+j], gg = gr[2*H+j], go = gr[3*H+j];
    float cn = sigmoidf_(gf) * cin[i] + sigmoidf_(gi) * tanhf(gg);
    cout[i] = cn;
    hout[i] = sigmoidf_(go) * tanhf(cn);
  }
}

__global__ void k_copy(const float* __restrict__ s, float* __restrict__ d, int n) {
  int i = blockIdx.x*256 + threadIdx.x;
  if (i < n) d[i] = s[i];
}

__global__ void k_argmax_rows(const float* __restrict__ x, int N, int* __restrict__ idx) {
  __shared__ float sv[256]; __shared__ int si[256];
  int r = blockIdx.x, t = threadIdx.x;
  const float* row = x + (size_t)r * N;
  float bv = -INFINITY; int bi = 0x7fffffff;
  for (int i = t; i < N; i += 256) {
    float v = row[i];
    if (v > bv || (v == bv && i < bi)) { bv = v; bi = i; }
  }
  sv[t]=bv; si[t]=bi; __syncthreads();
  for (int s = 128; s > 0; s >>= 1) {
    if (t < s) {
      if (sv[t+s] > sv[t] || (sv[t+s] == sv[t] && si[t+s] < si[t])) { sv[t]=sv[t+s]; si[t]=si[t+s]; }
    }
    __syncthreads();
  }
  if (t == 0) idx[r] = si[0];
}

__global__ void k_nstack(const float* __restrict__ noun, const float* __restrict__ verb,
                         const int* __restrict__ nidx, const int* __restrict__ vidx,
                         float* __restrict__ st) {
  int i = blockIdx.x*256 + threadIdx.x;
  if (i < BB*2*HSZ) {
    int m = i/(2*HSZ), k = i%(2*HSZ);
    st[i] = (k < HSZ) ? noun[(size_t)nidx[m]*HSZ + k] : verb[(size_t)vidx[m]*HSZ + (k-HSZ)];
  }
}

__global__ void k_mask_apply(const int* __restrict__ prev, const float* __restrict__ nh,
                             const float* __restrict__ nc, const float* __restrict__ nstk,
                             float* hint, float* cint, float* iout, float* istk) {
  int i = blockIdx.x*256 + threadIdx.x;
  if (i < BB*2*HSZ) {
    int m = i/(2*HSZ), k = i%(2*HSZ);
    int p = prev[m];
    bool msk = (p == 4) || (p == 5) || (p == 6);
    if (msk) {
      istk[i] = nstk[i];
      if (k < HSZ) { int j = m*HSZ + k; hint[j] = nh[j]; cint[j] = nc[j]; iout[j] = nh[j]; }
    }
  }
}

__global__ void k_softmax_rows(float* __restrict__ x, int N) {
  __shared__ float red[256];
  int r = blockIdx.x, t = threadIdx.x;
  float* row = x + (size_t)r * N;
  float mx = -INFINITY;
  for (int i = t; i < N; i += 256) mx = fmaxf(mx, row[i]);
  red[t] = mx; __syncthreads();
  for (int s = 128; s > 0; s >>= 1) { if (t < s) red[t] = fmaxf(red[t], red[t+s]); __syncthreads(); }
  mx = red[0]; __syncthreads();
  float sm = 0.0f;
  for (int i = t; i < N; i += 256) { float e = __expf(row[i]-mx); row[i] = e; sm += e; }
  red[t] = sm; __syncthreads();
  for (int s = 128; s > 0; s >>= 1) { if (t < s) red[t] += red[t+s]; __syncthreads(); }
  float inv = 1.0f / red[0];
  for (int i = t; i < N; i += 256) row[i] *= inv;
}

__global__ void k_logsoftmax_rows(float* __restrict__ x, int N) {
  __shared__ float red[256];
  int r = blockIdx.x, t = threadIdx.x;
  float* row = x + (size_t)r * N;
  float mx = -INFINITY;
  for (int i = t; i < N; i += 256) mx = fmaxf(mx, row[i]);
  red[t] = mx; __syncthreads();
  for (int s = 128; s > 0; s >>= 1) { if (t < s) red[t] = fmaxf(red[t], red[t+s]); __syncthreads(); }
  mx = red[0]; __syncthreads();
  float sm = 0.0f;
  for (int i = t; i < N; i += 256) sm += __expf(row[i]-mx);
  red[t] = sm; __syncthreads();
  for (int s = 128; s > 0; s >>= 1) { if (t < s) red[t] += red[t+s]; __syncthreads(); }
  float lse = mx + __logf(red[0]);
  for (int i = t; i < N; i += 256) row[i] -= lse;
}

__global__ void k_attn_cc(const float* __restrict__ w, const float* __restrict__ enc,
                          float* __restrict__ cc) {
  int j = blockIdx.x*256 + threadIdx.x;
  if (j < HSZ) {
    float acc[BB];
#pragma unroll
    for (int m = 0; m < BB; m++) acc[m] = 0.0f;
    for (int s = 0; s < SLEN; s++) {
      float e = enc[(size_t)s*HSZ + j];
#pragma unroll
      for (int m = 0; m < BB; m++) acc[m] += w[m*SLEN + s] * e;
    }
#pragma unroll
    for (int m = 0; m < BB; m++) cc[m*HSZ + j] = acc[m];
  }
}

// global beam top-8 over [8, VSZ] with jax-style first-occurrence tie-break
__global__ void k_topk(const float* __restrict__ scores, const float* __restrict__ logp,
                       float* __restrict__ nscores, int* __restrict__ bidx, int* __restrict__ tok) {
  const int T = 256;
  __shared__ float sv[T*BB];
  __shared__ int   si[T*BB];
  int t = threadIdx.x;
  float bv[BB]; int bi[BB];
#pragma unroll
  for (int j = 0; j < BB; j++) { bv[j] = -INFINITY; bi[j] = 0x7fffffff; }
  float sc[BB];
#pragma unroll
  for (int b = 0; b < BB; b++) sc[b] = scores[b];
  const int total = BB * VSZ;
  for (int i = t; i < total; i += T) {
    int b = i / VSZ;
    float v = sc[b] + logp[i];
    if (v > bv[BB-1] || (v == bv[BB-1] && i < bi[BB-1])) {
      int p = BB-1;
      while (p > 0 && (v > bv[p-1] || (v == bv[p-1] && i < bi[p-1]))) {
        bv[p] = bv[p-1]; bi[p] = bi[p-1]; p--;
      }
      bv[p] = v; bi[p] = i;
    }
  }
#pragma unroll
  for (int j = 0; j < BB; j++) { sv[t*BB+j] = bv[j]; si[t*BB+j] = bi[j]; }
  __syncthreads();
  for (int stride = T/2; stride > 0; stride >>= 1) {
    if (t < stride) {
      float mv[BB]; int mi[BB];
      int a = 0, b2 = 0;
      float* av = &sv[t*BB];            int* ai = &si[t*BB];
      float* bvp = &sv[(t+stride)*BB];  int* bip = &si[(t+stride)*BB];
#pragma unroll
      for (int j = 0; j < BB; j++) {
        bool ta = (av[a] > bvp[b2]) || (av[a] == bvp[b2] && ai[a] < bip[b2]);
        if (ta) { mv[j] = av[a]; mi[j] = ai[a]; a++; }
        else    { mv[j] = bvp[b2]; mi[j] = bip[b2]; b2++; }
      }
#pragma unroll
      for (int j = 0; j < BB; j++) { av[j] = mv[j]; ai[j] = mi[j]; }
    }
    __syncthreads();
  }
  if (t < BB) {
    nscores[t] = sv[t];
    bidx[t] = si[t] / VSZ;
    tok[t]  = si[t] % VSZ;
  }
}

__global__ void k_beam_gather(const int* __restrict__ bidx, const int* __restrict__ tok,
                              const float* opn, const float* hdec, const float* cdec,
                              const float* hint, const float* cint, const float* iout,
                              const float* istk,
                              float* ops2, float* hdec2, float* cdec2, float* hint2, float* cint2,
                              float* iout2, float* istk2,
                              int* prev, float* outtoks) {
  int b = blockIdx.x, t = threadIdx.x;
  int src = bidx[b];
  for (int k = t; k < HSZ; k += 256) {
    ops2 [b*HSZ+k] = opn [src*HSZ+k];
    hdec2[b*HSZ+k] = hdec[src*HSZ+k];
    cdec2[b*HSZ+k] = cdec[src*HSZ+k];
    hint2[b*HSZ+k] = hint[src*HSZ+k];
    cint2[b*HSZ+k] = cint[src*HSZ+k];
    iout2[b*HSZ+k] = iout[src*HSZ+k];
  }
  for (int k = t; k < 2*HSZ; k += 256) istk2[b*2*HSZ+k] = istk[src*2*HSZ+k];
  if (t == 0) { prev[b] = tok[b]; outtoks[b] = (float)tok[b]; }
}

__global__ void k_final(const float* __restrict__ scores, float* __restrict__ dout) {
  int t = threadIdx.x;
  if (t < BB) dout[t] = scores[t];
}

// ==================================================================================
extern "C" void kernel_launch(void* const* d_in, const int* in_sizes, int n_in,
                              void* d_out, int out_size, void* d_ws, size_t ws_size,
                              hipStream_t stream) {
  (void)in_sizes; (void)n_in; (void)out_size; (void)ws_size;
  const int*   inp        = (const int*)  d_in[0];
  const float* encemb     = (const float*)d_in[1];
  const float* encF_Wih   = (const float*)d_in[2];
  const float* encF_Whh   = (const float*)d_in[3];
  const float* encF_b     = (const float*)d_in[4];
  const float* encB_Wih   = (const float*)d_in[5];
  const float* encB_Whh   = (const float*)d_in[6];
  const float* encB_b     = (const float*)d_in[7];
  const float* inter_Wih  = (const float*)d_in[8];
  const float* inter_Whh  = (const float*)d_in[9];
  const float* inter_b    = (const float*)d_in[10];
  const float* internoun  = (const float*)d_in[11];
  const float* interverb  = (const float*)d_in[12];
  const float* noungen_W  = (const float*)d_in[13];
  const float* noungen_b  = (const float*)d_in[14];
  const float* verbgen_W  = (const float*)d_in[15];
  const float* verbgen_b  = (const float*)d_in[16];
  const float* decemb     = (const float*)d_in[17];
  const float* dec_Wih    = (const float*)d_in[18];
  const float* dec_Whh    = (const float*)d_in[19];
  const float* dec_b      = (const float*)d_in[20];
  const float* gen_W      = (const float*)d_in[21];
  const float* gen_b      = (const float*)d_in[22];
  const float* linin      = (const float*)d_in[23];
  const float* linout     = (const float*)d_in[24];
  float* dout = (float*)d_out;

  // ---- workspace bump allocator (256B aligned) ----
  char* ws = (char*)d_ws;
  size_t off = 0;
  auto alloc = [&](size_t bytes) -> void* {
    void* p = ws + off;
    off = (off + bytes + 255) & ~(size_t)255;
    return p;
  };
  // bf16 weights (total ~102 MB; decoder-hot set ~82 MB -> L2-resident)
  bf16_t* wihF_bf     = (bf16_t*)alloc((size_t)1024*512*2);
  bf16_t* wihB_bf     = (bf16_t*)alloc((size_t)1024*512*2);
  bf16_t* interWih_bf = (bf16_t*)alloc((size_t)2048*1024*2);
  bf16_t* interWhh_bf = (bf16_t*)alloc((size_t)2048*512*2);
  bf16_t* noun_bf     = (bf16_t*)alloc((size_t)NVSZ*512*2);
  bf16_t* verb_bf     = (bf16_t*)alloc((size_t)VVSZ*512*2);
  bf16_t* decWih_bf   = (bf16_t*)alloc((size_t)2048*2048*2);
  bf16_t* decWhh_bf   = (bf16_t*)alloc((size_t)2048*512*2);
  bf16_t* genW_bf     = (bf16_t*)alloc((size_t)VSZ*512*2);
  bf16_t* linin_bf    = (bf16_t*)alloc((size_t)512*512*2);
  bf16_t* linout_bf   = (bf16_t*)alloc((size_t)512*1024*2);
  bf16_t* emb_bf      = (bf16_t*)alloc((size_t)SLEN*HSZ*2);
  // encoder buffers
  float* Xf   = (float*)alloc((size_t)SLEN*1024*4);
  float* Xb   = (float*)alloc((size_t)SLEN*1024*4);
  float* outF = (float*)alloc((size_t)SLEN*H2*4);
  float* outB = (float*)alloc((size_t)SLEN*H2*4);
  float* hF = (float*)alloc(H2*4); float* cF = (float*)alloc(H2*4);
  float* hB = (float*)alloc(H2*4); float* cB = (float*)alloc(H2*4);
  float*  encf = (float*) alloc((size_t)SLEN*HSZ*4);
  bf16_t* encb = (bf16_t*)alloc((size_t)SLEN*HSZ*2);
  // beam state banks (ping-pong)
  struct Bank { float *ops,*hdec,*cdec,*hint,*cint,*iout,*istk; } bank[2];
  for (int q = 0; q < 2; q++) {
    bank[q].ops  = (float*)alloc((size_t)BB*HSZ*4);
    bank[q].hdec = (float*)alloc((size_t)BB*HSZ*4);
    bank[q].cdec = (float*)alloc((size_t)BB*HSZ*4);
    bank[q].hint = (float*)alloc((size_t)BB*HSZ*4);
    bank[q].cint = (float*)alloc((size_t)BB*HSZ*4);
    bank[q].iout = (float*)alloc((size_t)BB*HSZ*4);
    bank[q].istk = (float*)alloc((size_t)BB*2*HSZ*4);
  }
  // step temporaries
  float* nh    = (float*)alloc((size_t)BB*HSZ*4);
  float* nc    = (float*)alloc((size_t)BB*HSZ*4);
  float* nstk  = (float*)alloc((size_t)BB*2*HSZ*4);
  float* g_int = (float*)alloc((size_t)BB*2048*4);
  float* g_dec = (float*)alloc((size_t)BB*2048*4);
  float* nlog  = (float*)alloc((size_t)BB*NVSZ*4);
  float* vlog  = (float*)alloc((size_t)BB*VVSZ*4);
  float* logits= (float*)alloc((size_t)BB*VSZ*4);
  float* q_    = (float*)alloc((size_t)BB*HSZ*4);
  float* attw  = (float*)alloc((size_t)BB*SLEN*4);
  float* ccv   = (float*)alloc((size_t)BB*HSZ*4);
  float* opn   = (float*)alloc((size_t)BB*HSZ*4);
  bf16_t* A0 = (bf16_t*)alloc((size_t)16*2048*2);
  bf16_t* A1 = (bf16_t*)alloc((size_t)16*2048*2);
  bf16_t* A2 = (bf16_t*)alloc((size_t)16*2048*2);
  int* prev   = (int*)alloc(BB*4);
  int* nidx   = (int*)alloc(BB*4);
  int* vidx   = (int*)alloc(BB*4);
  int* bidx   = (int*)alloc(BB*4);
  int* tok    = (int*)alloc(BB*4);
  float* scores = (float*)alloc(BB*4);

  const int T = 256;
  auto cdiv = [](long long a, long long b) -> int { return (int)((a + b - 1) / b); };
  auto conv = [&](const float* s, bf16_t* d, long long n) {
    k_f2bf<<<cdiv(n, T), T, 0, stream>>>(s, d, (int)n);
  };
  auto gemm = [&](const bf16_t* A, const bf16_t* W, const float* bias, const float* Cin,
                  float* D, int M, int N, int K, int act) {
    dim3 grid(cdiv(N/16, 8), cdiv(M, 16));
    k_wmma_gemm<<<grid, 256, (size_t)16*K*sizeof(bf16_t), stream>>>(A, W, bias, Cin, D, M, N, K, act);
  };

  // ---------- phase A: weight conversion + encoder ----------
  conv(encF_Wih, wihF_bf,   (long long)1024*512);
  conv(encB_Wih, wihB_bf,   (long long)1024*512);
  conv(inter_Wih, interWih_bf, (long long)2048*1024);
  conv(inter_Whh, interWhh_bf, (long long)2048*512);
  conv(noungen_W, noun_bf,  (long long)NVSZ*512);
  conv(verbgen_W, verb_bf,  (long long)VVSZ*512);
  conv(dec_Wih,  decWih_bf, (long long)2048*2048);
  conv(dec_Whh,  decWhh_bf, (long long)2048*512);
  conv(gen_W,    genW_bf,   (long long)VSZ*512);
  conv(linin,    linin_bf,  (long long)512*512);
  conv(linout,   linout_bf, (long long)512*1024);
  k_embed_bf<<<cdiv((long long)SLEN*HSZ, T), T, 0, stream>>>(encemb, inp, emb_bf, SLEN, HSZ);

  // hoisted encoder input projections: X = emb @ Wih^T + b  (WMMA, M=512)
  gemm(emb_bf, wihF_bf, encF_b, nullptr, Xf, SLEN, 1024, 512, 0);
  gemm(emb_bf, wihB_bf, encB_b, nullptr, Xb, SLEN, 1024, 512, 0);
  k_enc_scan<<<2, 256, 0, stream>>>(Xf, Xb, encF_Whh, encB_Whh, outF, outB, hF, cF, hB, cB);
  k_build_enc<<<cdiv((long long)SLEN*HSZ, T), T, 0, stream>>>(outF, outB, encf, encb);
  k_init_states<<<cdiv((long long)BB*HSZ, T), T, 0, stream>>>(hF, cF, hB, cB,
      bank[0].hdec, bank[0].cdec, bank[0].hint, bank[0].cint, bank[0].ops, prev, scores);

  // initial inter-LSTM step on zero input: gates = hint @ Whh^T + b
  k_pack1<<<cdiv(16*HSZ, T), T, 0, stream>>>(bank[0].hint, A0, HSZ);
  gemm(A0, interWhh_bf, inter_b, nullptr, g_int, BB, 2048, 512, 0);
  k_lstm_cell<<<cdiv((long long)BB*HSZ, T), T, 0, stream>>>(g_int, bank[0].cint,
                                                            bank[0].hint, bank[0].cint, HSZ);
  k_copy<<<cdiv((long long)BB*HSZ, T), T, 0, stream>>>(bank[0].hint, bank[0].iout, BB*HSZ);
  // nv_pick(interout)
  k_pack1<<<cdiv(16*HSZ, T), T, 0, stream>>>(bank[0].iout, A0, HSZ);
  gemm(A0, noun_bf, noungen_b, nullptr, nlog, BB, NVSZ, 512, 0);
  gemm(A0, verb_bf, verbgen_b, nullptr, vlog, BB, VVSZ, 512, 0);
  k_argmax_rows<<<BB, 256, 0, stream>>>(nlog, NVSZ, nidx);
  k_argmax_rows<<<BB, 256, 0, stream>>>(vlog, VVSZ, vidx);
  k_nstack<<<cdiv((long long)BB*2*HSZ, T), T, 0, stream>>>(internoun, interverb, nidx, vidx, bank[0].istk);

  // ---------- phase B: 48 decoder steps ----------
  for (int step = 0; step < MAXLEN; step++) {
    Bank& cur = bank[step & 1];
    Bank& nxt = bank[(step + 1) & 1];

    // inter-LSTM candidate: gates = [ops|interout] @ Wih^T + hint @ Whh^T + b
    k_pack2<<<cdiv(16*1024, T), T, 0, stream>>>(cur.ops, HSZ, cur.iout, HSZ, A0);
    gemm(A0, interWih_bf, inter_b, nullptr, g_int, BB, 2048, 1024, 0);
    k_pack1<<<cdiv(16*HSZ, T), T, 0, stream>>>(cur.hint, A1, HSZ);
    gemm(A1, interWhh_bf, nullptr, g_int, g_int, BB, 2048, 512, 0);
    k_lstm_cell<<<cdiv((long long)BB*HSZ, T), T, 0, stream>>>(g_int, cur.cint, nh, nc, HSZ);
    // nv_pick(nh)
    k_pack1<<<cdiv(16*HSZ, T), T, 0, stream>>>(nh, A2, HSZ);
    gemm(A2, noun_bf, noungen_b, nullptr, nlog, BB, NVSZ, 512, 0);
    gemm(A2, verb_bf, verbgen_b, nullptr, vlog, BB, VVSZ, 512, 0);
    k_argmax_rows<<<BB, 256, 0, stream>>>(nlog, NVSZ, nidx);
    k_argmax_rows<<<BB, 256, 0, stream>>>(vlog, VVSZ, vidx);
    k_nstack<<<cdiv((long long)BB*2*HSZ, T), T, 0, stream>>>(internoun, interverb, nidx, vidx, nstk);
    // conditional refresh on punctuation
    k_mask_apply<<<cdiv((long long)BB*2*HSZ, T), T, 0, stream>>>(prev, nh, nc, nstk,
        cur.hint, cur.cint, cur.iout, cur.istk);

    // decoder LSTM: gates = [decemb[prev]|ops|interstack] @ Wih^T + hdec @ Whh^T + b
    k_pack_dec<<<cdiv(16*2048, T), T, 0, stream>>>(decemb, prev, cur.ops, cur.istk, A0);
    gemm(A0, decWih_bf, dec_b, nullptr, g_dec, BB, 2048, 2048, 0);
    k_pack1<<<cdiv(16*HSZ, T), T, 0, stream>>>(cur.hdec, A1, HSZ);
    gemm(A1, decWhh_bf, nullptr, g_dec, g_dec, BB, 2048, 512, 0);
    k_lstm_cell<<<cdiv((long long)BB*HSZ, T), T, 0, stream>>>(g_dec, cur.cdec, cur.hdec, cur.cdec, HSZ);

    // attention: q = hdec @ linin^T ; w = softmax(q @ enc^T) ; cc = w @ enc
    k_pack1<<<cdiv(16*HSZ, T), T, 0, stream>>>(cur.hdec, A1, HSZ);
    gemm(A1, linin_bf, nullptr, nullptr, q_, BB, HSZ, 512, 0);
    k_pack1<<<cdiv(16*HSZ, T), T, 0, stream>>>(q_, A2, HSZ);
    gemm(A2, encb, nullptr, nullptr, attw, BB, SLEN, 512, 0);
    k_softmax_rows<<<BB, 256, 0, stream>>>(attw, SLEN);
    k_attn_cc<<<cdiv(HSZ, T), T, 0, stream>>>(attw, encf, ccv);
    // op = tanh([cc|hdec] @ linout^T)
    k_pack2<<<cdiv(16*1024, T), T, 0, stream>>>(ccv, HSZ, cur.hdec, HSZ, A0);
    gemm(A0, linout_bf, nullptr, nullptr, opn, BB, HSZ, 1024, 1);

    // generator logits + log-softmax (the bandwidth monster: bf16 WMMA from L2)
    k_pack1<<<cdiv(16*HSZ, T), T, 0, stream>>>(opn, A1, HSZ);
    gemm(A1, genW_bf, gen_b, nullptr, logits, BB, VSZ, 512, 0);
    k_logsoftmax_rows<<<BB, 256, 0, stream>>>(logits, VSZ);

    // global beam top-8, then gather states into next bank
    k_topk<<<1, 256, 0, stream>>>(scores, logits, scores, bidx, tok);
    k_beam_gather<<<BB, 256, 0, stream>>>(bidx, tok, opn, cur.hdec, cur.cdec,
        cur.hint, cur.cint, cur.iout, cur.istk,
        nxt.ops, nxt.hdec, nxt.cdec, nxt.hint, nxt.cint, nxt.iout, nxt.istk,
        prev, dout + BB + (size_t)step * BB);
  }

  k_final<<<1, 32, 0, stream>>>(scores, dout);
}
